// MyRNN_21414706938125
// MI455X (gfx1250) — compile-verified, matrix operand order checked by reading, not run on hardware
//
#include <hip/hip_runtime.h>
#include <hip/hip_bf16.h>
#include <math.h>

typedef __attribute__((ext_vector_type(16))) __bf16 v16bf;
typedef __attribute__((ext_vector_type(8)))  float  v8f;

#define Bsz   32
#define Tlen  1000
#define Din   1280
#define Hdim  32
#define G3    96          // 3*H gates
#define OUTD  5
#define MROWS (Bsz * Tlen)

// ---------------------------------------------------------------------------
// Fragment gather: 16-bit A-matrix 16x32 (MxK) layout per CDNA5 ISA 7.12.2.
// lane&15 selects the tile row; lane>>4 selects the K half (+8 offset).
// VGPR v holds K pair { (v<4 ? 2v : 16+2(v-4)) + 8*kh , +1 }.
// B (KxN) uses the identical gather applied to B^T rows; W_ih is stored
// [N,K] row-major == B^T, so one helper serves both operands.
// ---------------------------------------------------------------------------
__device__ __forceinline__ v16bf load_frag_bf16_from_f32(
    const float* __restrict__ rowptr, int k0, int kh) {
  v16bf f;
#pragma unroll
  for (int v = 0; v < 8; ++v) {
    const int kk = k0 + ((v < 4) ? (2 * v) : (16 + 2 * (v - 4))) + 8 * kh;
    f[2 * v]     = (__bf16)rowptr[kk];
    f[2 * v + 1] = (__bf16)rowptr[kk + 1];
  }
  return f;
}

// Fused fwd+bwd input projection:
//   Df = A @ Bf^T + biasf ; Db = A @ Bb^T + biasb   (N = 96 each)
// One wave owns a 32x16 output tile (two 16x16 WMMA tiles sharing one B
// fragment -> halves weight traffic, 2 independent WMMA chains per k-step).
// 12 N-tiles total: tn<6 -> forward weights, tn>=6 -> backward weights.
template <int K>
__global__ void wmma_gemm_bias2(const float* __restrict__ A, int lda,
                                const float* __restrict__ Bf,
                                const float* __restrict__ Bb, int ldb,
                                const float* __restrict__ biasf,
                                const float* __restrict__ biasb,
                                float* __restrict__ Df, float* __restrict__ Db,
                                int ldd, int M) {
  const int wave = blockIdx.x * (blockDim.x >> 5) + (threadIdx.x >> 5);
  const int tmb  = wave / 12;          // 32-row M block
  int       tn   = wave - tmb * 12;    // 0..11
  const int m0   = tmb * 32;
  if (m0 >= M) return;

  const float* Bw   = Bf;
  const float* bias = biasf;
  float*       Dst  = Df;
  if (tn >= 6) { Bw = Bb; bias = biasb; Dst = Db; tn -= 6; }

  const int lane = threadIdx.x & 31;
  const int r    = lane & 15;   // tile row (A) / tile col (B, D)
  const int kh   = lane >> 4;   // K-half select

  const float* arow0 = A  + (size_t)(m0 + r) * lda;
  const float* arow1 = A  + (size_t)(m0 + 16 + r) * lda;
  const float* brow  = Bw + (size_t)(tn * 16 + r) * ldb;

  const float bv = bias[tn * 16 + r];   // lane -> N column for C/D layout
  v8f c0, c1;
#pragma unroll
  for (int i = 0; i < 8; ++i) { c0[i] = bv; c1[i] = bv; }

#pragma unroll 2
  for (int k0 = 0; k0 < K; k0 += 32) {
    v16bf b  = load_frag_bf16_from_f32(brow,  k0, kh);
    v16bf a0 = load_frag_bf16_from_f32(arow0, k0, kh);
    v16bf a1 = load_frag_bf16_from_f32(arow1, k0, kh);
    // (neg_a, A, neg_b, B, c_mod, C, reuse_a, reuse_b)
    c0 = __builtin_amdgcn_wmma_f32_16x16x32_bf16(false, a0, false, b,
                                                 (short)0, c0, false, false);
    c1 = __builtin_amdgcn_wmma_f32_16x16x32_bf16(false, a1, false, b,
                                                 (short)0, c1, false, false);
  }

  // C/D layout: VGPR v, lanes 0-15 -> M=v, N=lane; lanes 16-31 -> M=v+8.
#pragma unroll
  for (int v = 0; v < 8; ++v) {
    Dst[(size_t)(m0 + v + 8 * kh) * ldd + tn * 16 + r]      = c0[v];
    Dst[(size_t)(m0 + 16 + v + 8 * kh) * ldd + tn * 16 + r] = c1[v];
  }
}

// ---------------------------------------------------------------------------
// Persistent-block GRU recurrence. blockIdx.x = direction (0 fwd, 1 bwd).
// 1024 threads: thread (b,j) owns h[b][j]. W_hh and h live in LDS.
// h is double-buffered -> single barrier per timestep (critical-path latency).
// PyTorch gate math: r,z = sigmoid(gx + h@Whh^T + bhh); n = tanh(gn + r*hn).
// ---------------------------------------------------------------------------
__device__ __forceinline__ float sigmoidf_(float x) {
  return 1.0f / (1.0f + __expf(-x));
}

__global__ void gru_scan(const float* __restrict__ gxf, const float* __restrict__ gxb,
                         const float* __restrict__ Wf,  const float* __restrict__ Wb,
                         const float* __restrict__ bf,  const float* __restrict__ bb,
                         float* __restrict__ out) {
  const int dir       = blockIdx.x;
  const float* gx     = dir ? gxb : gxf;
  const float* W      = dir ? Wb  : Wf;
  const float* bh     = dir ? bb  : bf;
  const int    coloff = dir * Hdim;

  __shared__ float sW[G3][Hdim + 1];       // +1 pad: kill bank conflicts
  __shared__ float sB[G3];
  __shared__ float sh[2][Bsz][Hdim + 1];   // double-buffered hidden state

  const int tid = threadIdx.x;
  const int b   = tid >> 5;
  const int j   = tid & 31;

  for (int i = tid; i < G3 * Hdim; i += blockDim.x) sW[i >> 5][i & 31] = W[i];
  if (tid < G3) sB[tid] = bh[tid];
  sh[0][b][j] = 0.0f;
  __syncthreads();

  int p = 0;
  for (int step = 0; step < Tlen; ++step) {
    const int t = dir ? (Tlen - 1 - step) : step;
    const float* g = gx + ((size_t)b * Tlen + t) * G3;

    // Prefetch next timestep's gate row while we compute (global_prefetch_b8)
    if (step + 1 < Tlen) {
      const int tn = dir ? (t - 1) : (t + 1);
      __builtin_prefetch(gx + ((size_t)b * Tlen + tn) * G3 + j, 0, 0);
    }

    float hr = sB[j], hz = sB[j + 32], hn = sB[j + 64];
#pragma unroll 8
    for (int k = 0; k < Hdim; ++k) {
      const float hk = sh[p][b][k];
      hr += hk * sW[j][k];
      hz += hk * sW[j + 32][k];
      hn += hk * sW[j + 64][k];
    }

    const float rr  = sigmoidf_(g[j]      + hr);
    const float zz  = sigmoidf_(g[j + 32] + hz);
    const float nn  = tanhf    (g[j + 64] + rr * hn);
    const float hnw = (1.0f - zz) * nn + zz * sh[p][b][j];

    sh[p ^ 1][b][j] = hnw;   // write only the other buffer: no race with reads
    out[((size_t)b * Tlen + t) * (2 * Hdim) + coloff + j] = hnw;
    __syncthreads();         // one barrier per step
    p ^= 1;
  }
}

// y[bt,o] = h[bt,:] . fc_w[o,:] + fc_b[o]
__global__ void fc_kernel(const float* __restrict__ h, const float* __restrict__ fw,
                          const float* __restrict__ fb, float* __restrict__ y) {
  const int idx = blockIdx.x * blockDim.x + threadIdx.x;
  if (idx >= MROWS * OUTD) return;
  const int bt = idx / OUTD, o = idx - bt * OUTD;
  const float* hr = h  + (size_t)bt * (2 * Hdim);
  const float* w  = fw + o * (2 * Hdim);
  float acc = fb[o];
#pragma unroll
  for (int k = 0; k < 2 * Hdim; ++k) acc += hr[k] * w[k];
  y[idx] = acc;
}

extern "C" void kernel_launch(void* const* d_in, const int* in_sizes, int n_in,
                              void* d_out, int out_size, void* d_ws, size_t ws_size,
                              hipStream_t stream) {
  const float* x        = (const float*)d_in[0];
  const float* w_ih_l0  = (const float*)d_in[1];
  const float* w_hh_l0  = (const float*)d_in[2];
  const float* b_ih_l0  = (const float*)d_in[3];
  const float* b_hh_l0  = (const float*)d_in[4];
  const float* w_ih_l0r = (const float*)d_in[5];
  const float* w_hh_l0r = (const float*)d_in[6];
  const float* b_ih_l0r = (const float*)d_in[7];
  const float* b_hh_l0r = (const float*)d_in[8];
  const float* w_ih_l1  = (const float*)d_in[9];
  const float* w_hh_l1  = (const float*)d_in[10];
  const float* b_ih_l1  = (const float*)d_in[11];
  const float* b_hh_l1  = (const float*)d_in[12];
  const float* w_ih_l1r = (const float*)d_in[13];
  const float* w_hh_l1r = (const float*)d_in[14];
  const float* b_ih_l1r = (const float*)d_in[15];
  const float* b_hh_l1r = (const float*)d_in[16];
  const float* fc_w     = (const float*)d_in[17];
  const float* fc_b     = (const float*)d_in[18];

  // Workspace: gx_f | gx_b | h0   (gx buffers reused by layer 1) ~33 MB
  float* ws   = (float*)d_ws;
  float* gx_f = ws;
  float* gx_b = gx_f + (size_t)MROWS * G3;
  float* h0   = gx_b + (size_t)MROWS * G3;

  // Output: y [32000*5] then h [32000*64]
  float* y    = (float*)d_out;
  float* hout = y + (size_t)MROWS * OUTD;

  // (MROWS/32) M-blocks x 12 N-tiles, 4 waves per 128-thread WG
  const int waves  = (MROWS / 32) * 12;   // 12000
  const int blocks = waves / 4;           // 3000

  // Layer 0 input projections, fwd+bwd fused (K = 1280: bulk of all FLOPs)
  wmma_gemm_bias2<Din><<<blocks, 128, 0, stream>>>(
      x, Din, w_ih_l0, w_ih_l0r, Din, b_ih_l0, b_ih_l0r, gx_f, gx_b, G3, MROWS);
  gru_scan<<<2, 1024, 0, stream>>>(gx_f, gx_b, w_hh_l0, w_hh_l0r, b_hh_l0, b_hh_l0r, h0);

  // Layer 1 input projections (K = 64)
  wmma_gemm_bias2<2 * Hdim><<<blocks, 128, 0, stream>>>(
      h0, 2 * Hdim, w_ih_l1, w_ih_l1r, 2 * Hdim, b_ih_l1, b_ih_l1r, gx_f, gx_b, G3, MROWS);
  gru_scan<<<2, 1024, 0, stream>>>(gx_f, gx_b, w_hh_l1, w_hh_l1r, b_hh_l1, b_hh_l1r, hout);

  fc_kernel<<<(MROWS * OUTD + 255) / 256, 256, 0, stream>>>(hout, fc_w, fc_b, y);

  (void)in_sizes; (void)n_in; (void)out_size; (void)ws_size;
}
